// CasualSelfAttention_17652315587092
// MI455X (gfx1250) — matrix operand order repaired and checked by reading, not compiled, not run on hardware
//
#include <hip/hip_runtime.h>
#include <hip/hip_bf16.h>
#include <stddef.h>

typedef __bf16 bf16;
typedef bf16  v16bf __attribute__((ext_vector_type(16)));
typedef bf16  v8bf  __attribute__((ext_vector_type(8)));
typedef bf16  v4bf  __attribute__((ext_vector_type(4)));
typedef float v8f   __attribute__((ext_vector_type(8)));
typedef unsigned int u32x4 __attribute__((ext_vector_type(4)));
typedef int          i32x4 __attribute__((ext_vector_type(4)));
typedef int          i32x8 __attribute__((ext_vector_type(8)));

#define WMMA_BF16(a, b, c) \
    __builtin_amdgcn_wmma_f32_16x16x32_bf16(false, (a), false, (b), (short)0, (c), false, false)

#if __has_builtin(__builtin_amdgcn_tensor_load_to_lds)
#define HAVE_TDM 1
#endif

struct MaskOn  { static constexpr bool value = true;  };
struct MaskOff { static constexpr bool value = false; };

// ISA 7.12.2 fragment K-index maps (wave32, 16-bit data)
__device__ __forceinline__ v16bf frag_cat(v8bf lo, v8bf hi8) {
    return __builtin_shufflevector(lo, hi8, 0, 1, 2, 3, 4, 5, 6, 7,
                                            8, 9, 10, 11, 12, 13, 14, 15);
}
__device__ __forceinline__ v16bf load_fragA(const bf16* p, int hi) {
    return frag_cat(*(const v8bf*)(p + 8 * hi), *(const v8bf*)(p + 16 + 8 * hi));
}
__device__ __forceinline__ v16bf load_fragB(const bf16* p, int hi) {
    return frag_cat(*(const v8bf*)(p + 8 * hi), *(const v8bf*)(p + 16 + 8 * hi));
}

#ifdef HAVE_TDM
// TDM 2D tile load: 128 rows x 32 bf16, row stride = stride_elems, into LDS
// at lds_off; hardware pad (4 dwords after every 16 dwords) -> 40-half rows.
__device__ __forceinline__ void tdm_load_tile(unsigned lds_off, const bf16* gp,
                                              int stride_elems) {
    unsigned long long ga = (unsigned long long)(const void*)gp;
    u32x4 g0;
    g0[0] = 1u;                                   // count = 1 valid descriptor
    g0[1] = lds_off;                              // lds_addr
    g0[2] = (unsigned)ga;                         // global_addr[31:0]
    g0[3] = (unsigned)(ga >> 32) | (2u << 30);    // global_addr[56:32] | type=2
    i32x8 g1;
    g1[0] = (1 << 16) | (1 << 20) | (3 << 22) | (3 << 25); // dsz=2B, pad 16dw->4dw
    g1[1] = (32 << 16);                           // tensor_dim0 = 32
    g1[2] = (128 << 16);                          // tensor_dim1 = 128
    g1[3] = (32 << 16);                           // tile_dim0 = 32
    g1[4] = 128;                                  // tile_dim1 = 128
    g1[5] = stride_elems;                         // tensor_dim0_stride
    g1[6] = 0; g1[7] = 0;
    i32x4 z4 = {0, 0, 0, 0};
#if __clang_major__ >= 23
    i32x8 z8 = {0, 0, 0, 0, 0, 0, 0, 0};
    __builtin_amdgcn_tensor_load_to_lds(g0, g1, z4, z4, z8, 0);
#else
    __builtin_amdgcn_tensor_load_to_lds(g0, g1, z4, z4, 0);
#endif
}
#endif

// ---------------------------------------------------------------------------
// GEMM: Out[M,N] = A[M][K](bf16) @ BT[N][K](bf16)^T + bias[n]
// block = 256 threads (8 waves), tile 128x128, K-step 32; waves 4(M)x2(N),
// 8 wmma each per k-step. Double-buffered TDM pipeline: wave 0 issues tile
// i+1 DMAs then s_wait_tensorcnt(2) so tile i is complete while i+1 streams.
// ---------------------------------------------------------------------------
template <bool OUT_BF16>
__global__ void __launch_bounds__(256)
gemm_bf16_kernel(const bf16* __restrict__ A, const bf16* __restrict__ BT,
                 const float* __restrict__ bias, void* __restrict__ Out,
                 bf16* __restrict__ OutT, int K, int ldo)
{
    __shared__ __align__(16) bf16 smem[2][2][128 * 40];   // [buf][A/B][row][k]

    const int t    = threadIdx.x;
    const int lane = t & 31;
    const int hi   = lane >> 4;
    const int ln   = lane & 15;
    const int wm   = (t >> 5) >> 1;   // 0..3
    const int wn   = (t >> 5) & 1;    // 0..1
    const int m0   = blockIdx.y * 128;
    const int n0   = blockIdx.x * 128;

    v8f acc[2][4] = {};

    auto compute = [&](const bf16* Asb, const bf16* Bsb) {
        v16bf af[2];
#pragma unroll
        for (int ma = 0; ma < 2; ++ma)
            af[ma] = load_fragA(&Asb[(wm * 32 + ma * 16 + ln) * 40], hi);
#pragma unroll
        for (int bn = 0; bn < 4; ++bn) {
            v16bf bfrag = load_fragB(&Bsb[(wn * 64 + bn * 16 + ln) * 40], hi);
#pragma unroll
            for (int ma = 0; ma < 2; ++ma)
                acc[ma][bn] = WMMA_BF16(af[ma], bfrag, acc[ma][bn]);
        }
    };

    const int nIter = K >> 5;
#ifdef HAVE_TDM
    if (t < 32) {
        tdm_load_tile((unsigned)(size_t)(void*)&smem[0][0][0], &A[m0 * K], K);
        tdm_load_tile((unsigned)(size_t)(void*)&smem[0][1][0], &BT[n0 * K], K);
    }
    for (int it = 0; it < nIter; ++it) {
        const int cur = it & 1;
        if (t < 32) {
            if (it + 1 < nIter) {
                int k1 = (it + 1) << 5;
                tdm_load_tile((unsigned)(size_t)(void*)&smem[cur ^ 1][0][0],
                              &A[m0 * K + k1], K);
                tdm_load_tile((unsigned)(size_t)(void*)&smem[cur ^ 1][1][0],
                              &BT[n0 * K + k1], K);
                __builtin_amdgcn_s_wait_tensorcnt(2);   // tile `it` complete
            } else {
                __builtin_amdgcn_s_wait_tensorcnt(0);
            }
        }
        __syncthreads();
        compute(&smem[cur][0][0], &smem[cur][1][0]);
        __syncthreads();   // waves done reading before buf is re-DMA'd
    }
#else
    for (int it = 0; it < nIter; ++it) {
        int k0 = it << 5;
#pragma unroll
        for (int e = 0; e < 2; ++e) {      // plain bf16 tile copy fallback
            int i = t + 256 * e;           // 0..511 chunks of 8 halves
            int r = i >> 2, c8 = (i & 3) * 8;
            *(v8bf*)&smem[0][0][r * 40 + c8] = *(const v8bf*)&A[(m0 + r) * K + k0 + c8];
            *(v8bf*)&smem[0][1][r * 40 + c8] = *(const v8bf*)&BT[(n0 + r) * K + k0 + c8];
        }
        __syncthreads();
        compute(&smem[0][0][0], &smem[0][1][0]);
        __syncthreads();
    }
#endif

    // C/D layout: lane, vgpr v -> m = v + 8*hi, n = ln
#pragma unroll
    for (int ma = 0; ma < 2; ++ma) {
#pragma unroll
        for (int bn = 0; bn < 4; ++bn) {
            int col  = n0 + wn * 64 + bn * 16 + ln;
            float bv = bias[col];
            int r0   = m0 + wm * 32 + ma * 16 + 8 * hi;
            v8bf tv;
#pragma unroll
            for (int v = 0; v < 8; ++v) {
                float val = acc[ma][bn][v] + bv;
                tv[v] = (bf16)val;
                if (OUT_BF16)
                    ((bf16*)Out)[(r0 + v) * ldo + col] = (bf16)val;
                else
                    ((float*)Out)[(r0 + v) * ldo + col] = val;
            }
            if (OUT_BF16 && OutT) {     // OutT[b,h,d,t], 8 rows = 8 consecutive t
                int bb = r0 >> 11, t0 = r0 & 2047;
                int idxT = (((bb << 4) + (col >> 6)) << 17) + ((col & 63) << 11) + t0;
                *(v8bf*)&OutT[idxT] = tv;
            }
        }
    }
}

// ---------------------------------------------------------------------------
// Elementwise fp32 -> bf16 (x tensor), one float4 per thread
// ---------------------------------------------------------------------------
__global__ void __launch_bounds__(256)
conv_bf16_kernel(const float* __restrict__ X, bf16* __restrict__ Xb)
{
    int i = blockIdx.x * 256 + threadIdx.x;
    float4 fv = ((const float4*)X)[i];
    v4bf s = { (bf16)fv.x, (bf16)fv.y, (bf16)fv.z, (bf16)fv.w };
    ((v4bf*)Xb)[i] = s;
}

// ---------------------------------------------------------------------------
// Weight convert + transpose: WT[n][k] = bf16(W[k][col0+n]); 64x64 LDS tiles
// ---------------------------------------------------------------------------
__global__ void __launch_bounds__(256)
convT_bf16_kernel(const float* __restrict__ W, bf16* __restrict__ WT,
                  int ldw, int col0, int Kdim)
{
    __shared__ float Lt[64 * 65];
    const int t  = threadIdx.x;
    const int k0 = blockIdx.x * 64;
    const int n0 = blockIdx.y * 64;
#pragma unroll
    for (int e = 0; e < 16; ++e) {
        int i = t + 256 * e;
        int r = i >> 6, c = i & 63;
        Lt[c * 65 + r] = W[(k0 + r) * ldw + col0 + n0 + c];
    }
    __syncthreads();
#pragma unroll
    for (int e = 0; e < 4; ++e) {
        int j  = t + 256 * e;
        int n  = j >> 4, k4 = (j & 15) * 4;
        const float* p = &Lt[n * 65 + k4];
        v4bf s = { (bf16)p[0], (bf16)p[1], (bf16)p[2], (bf16)p[3] };
        *(v4bf*)&WT[(n0 + n) * Kdim + k0 + k4] = s;
    }
}

// ---------------------------------------------------------------------------
// Causal flash attention, q=k=v. One wave (block=32) per (b,h,32-row Q tile).
// ---------------------------------------------------------------------------
__global__ void __launch_bounds__(32)
attn_kernel(const bf16* __restrict__ kbf, const bf16* __restrict__ kbfT,
            bf16* __restrict__ ybf, int T, int C, int H)
{
    const int hd   = 64;
    const int lane = threadIdx.x;
    const int hi   = lane >> 4;
    const int ln   = lane & 15;

    int idx      = blockIdx.x;
    const int TQ = T >> 5;
    const int qi = idx % TQ; idx /= TQ;
    const int h  = idx % H;  idx /= H;
    const int b  = idx;
    const int qbase = qi * 32;

    __shared__ float Ssm[32 * 33];
    __shared__ __align__(16) bf16 Psm[32 * 40];
    __shared__ float mrow[32], lrow[32], arow[32], linv[32];

    // force block-uniform bases into SGPRs
    const bf16* base  = kbf + __builtin_amdgcn_readfirstlane(b * (T * C) + h * hd);
    const bf16* baseT = kbfT + __builtin_amdgcn_readfirstlane(((b * H + h) << 6) * T);

    v16bf qf[2][2];
#pragma unroll
    for (int qa = 0; qa < 2; ++qa)
#pragma unroll
        for (int s = 0; s < 2; ++s) {
            v16bf q = load_fragA(base + (qbase + qa * 16 + ln) * C + s * 32, hi);
#pragma unroll
            for (int e = 0; e < 16; ++e)
                q[e] = (bf16)((float)q[e] * 0.125f);      // fold 1/sqrt(64)
            qf[qa][s] = q;
        }

    v8f o[2][4] = {};
    mrow[lane] = -1e30f;
    lrow[lane] = 0.f;
    __syncthreads();

    auto chunk = [&](int j0, auto maskc) {
        constexpr bool MASK = decltype(maskc)::value;
#pragma unroll
        for (int jt = 0; jt < 2; ++jt) {
            v16bf kf[2];
#pragma unroll
            for (int s = 0; s < 2; ++s)
                kf[s] = load_fragB(base + (j0 + jt * 16 + ln) * C + s * 32, hi);
#pragma unroll
            for (int qa = 0; qa < 2; ++qa) {
                v8f sacc = {};
                sacc = WMMA_BF16(qf[qa][0], kf[0], sacc);
                sacc = WMMA_BF16(qf[qa][1], kf[1], sacc);
#pragma unroll
                for (int v = 0; v < 8; ++v) {
                    int m = qa * 16 + v + 8 * hi;
                    float sv = sacc[v];
                    if (MASK) {
                        int gq = qbase + m, gk = j0 + jt * 16 + ln;
                        if (gk > gq) sv = -1e30f;
                    }
                    Ssm[m * 33 + jt * 16 + ln] = sv;
                }
            }
        }
        __syncthreads();

        {   // online softmax: each lane owns one of the 32 rows
            const int m = lane;
            float mx = mrow[m];
            for (int c = 0; c < 32; ++c) mx = fmaxf(mx, Ssm[m * 33 + c]);
            float alpha = __expf(mrow[m] - mx);
            float lsum  = lrow[m] * alpha;
            for (int c = 0; c < 32; ++c) {
                float p = __expf(Ssm[m * 33 + c] - mx);
                Psm[m * 40 + c] = (bf16)p;
                lsum += p;
            }
            mrow[m] = mx; lrow[m] = lsum; arow[m] = alpha;
        }
        __syncthreads();

#pragma unroll
        for (int dt = 0; dt < 4; ++dt) {
            v16bf vf = load_fragB(baseT + ((dt * 16 + ln) << 11) + j0, hi); // T=2048
#pragma unroll
            for (int qa = 0; qa < 2; ++qa) {
                v8f c = o[qa][dt];
#pragma unroll
                for (int v = 0; v < 8; ++v) c[v] *= arow[qa * 16 + v + 8 * hi];
                v16bf pf = load_fragA(&Psm[(qa * 16 + ln) * 40], hi);
                o[qa][dt] = WMMA_BF16(pf, vf, c);
            }
        }
        __syncthreads();
    };

    for (int j0 = 0; j0 < qbase; j0 += 32) chunk(j0, MaskOff{});
    chunk(qbase, MaskOn{});

    linv[lane] = 1.0f / lrow[lane];   // one rcp per row
    __syncthreads();

#pragma unroll
    for (int qa = 0; qa < 2; ++qa)
#pragma unroll
        for (int dt = 0; dt < 4; ++dt)
#pragma unroll
            for (int v = 0; v < 8; ++v) {
                int m = qa * 16 + v + 8 * hi;
                ybf[(b * T + qbase + m) * C + h * hd + dt * 16 + ln] =
                    (bf16)(o[qa][dt][v] * linv[m]);
            }
}

// ---------------------------------------------------------------------------
extern "C" void kernel_launch(void* const* d_in, const int* in_sizes, int n_in,
                              void* d_out, int out_size, void* d_ws, size_t ws_size,
                              hipStream_t stream)
{
    const float* x      = (const float*)d_in[0];
    const float* W_attn = (const float*)d_in[1];
    const float* b_attn = (const float*)d_in[2];
    const float* W_proj = (const float*)d_in[3];
    const float* b_proj = (const float*)d_in[4];
    float* out = (float*)d_out;

    const int Bsz = 4, T = 2048, C = 1024, H = 16;
    const int M = Bsz * T;                       // 8192

    const size_t eb = (size_t)M * C * sizeof(bf16);   // 16 MB
    char* ws = (char*)d_ws;
    bf16* kbf  = (bf16*)(ws);                    // K projection (bf16)
    bf16* ybf  = (bf16*)(ws + eb);               // attention out (bf16)
    bf16* kbfT = (bf16*)(ws + 2 * eb);           // K proj transposed [b,h,d,t]
    bf16* xbf  = (bf16*)(ws + 3 * eb);           // x in bf16
    bf16* WkT  = (bf16*)(ws + 4 * eb);           // W_attn K-slice^T [n][k]
    bf16* WpT  = (bf16*)(ws + 4 * eb + (size_t)C * C * sizeof(bf16));

    // 0) one-time conversions (bandwidth-trivial at 23.3 TB/s)
    conv_bf16_kernel<<<dim3(M * C / 4 / 256), dim3(256), 0, stream>>>(x, xbf);
    convT_bf16_kernel<<<dim3(C / 64, C / 64), dim3(256), 0, stream>>>(
        W_attn, WkT, 3 * C, C, C);
    convT_bf16_kernel<<<dim3(C / 64, C / 64), dim3(256), 0, stream>>>(
        W_proj, WpT, C, 0, C);

    dim3 blk(256);
    dim3 g_gemm(C / 128, M / 128);

    // 1) K projection -> kbf (+ transposed copy for V fragments)
    gemm_bf16_kernel<true><<<g_gemm, blk, 0, stream>>>(
        xbf, WkT, b_attn + C, kbf, kbfT, /*K=*/C, /*ldo=*/C);

    // 2) causal self-attention with q=k=v -> ybf (bf16)
    dim3 g_attn(Bsz * H * (T / 32));
    attn_kernel<<<g_attn, dim3(32), 0, stream>>>(kbf, kbfT, ybf, T, C, H);

    // 3) output projection -> fp32 out
    gemm_bf16_kernel<false><<<g_gemm, blk, 0, stream>>>(
        ybf, WpT, b_proj, out, nullptr, /*K=*/C, /*ldo=*/C);
}
